// VectorQuantizer_83056077571008
// MI455X (gfx1250) — matrix operand order repaired and checked by reading, not compiled, not run on hardware
//
#include <hip/hip_runtime.h>

typedef __attribute__((ext_vector_type(16))) __bf16 v16bf;
typedef __attribute__((ext_vector_type(8)))  __bf16 v8bf;
typedef __attribute__((ext_vector_type(8)))  float  v8f;

#define NUM_EMB   1024
#define DIM       256
#define N_TOTAL   32768        // 32 * 32 * 32 rows
#define NTILE     16
#define N_ELEMS   8388608      // 32*256*32*32
#define LOSS_SCALE (1.25f / 8388608.0f)

// ---------------------------------------------------------------------------
// Prep: codebook f32 -> bf16, per-row squared norms, zero the loss slot.
// grid = 1024 blocks (one per codebook row), block = 256 threads (one per c).
// ---------------------------------------------------------------------------
__global__ void vq_prep_kernel(const float* __restrict__ cb,
                               unsigned short* __restrict__ cb16_raw,
                               float* __restrict__ enorm,
                               float* __restrict__ loss_slot) {
    __bf16* cb16 = (__bf16*)cb16_raw;
    int k = blockIdx.x;
    int c = threadIdx.x;
    float e = cb[(size_t)k * DIM + c];
    cb16[(size_t)k * DIM + c] = (__bf16)e;

    __shared__ float s[256];
    s[c] = e * e;
    __syncthreads();
    #pragma unroll
    for (int off = 128; off > 0; off >>= 1) {
        if (c < off) s[c] += s[c + off];
        __syncthreads();
    }
    if (c == 0) enorm[k] = s[0];
    if (k == 0 && c == 0) *loss_slot = 0.0f;
}

// ---------------------------------------------------------------------------
// Main: per-workgroup 16-row tile of x_ (N x C), full K=1024 argmin via
// v_wmma_f32_16x16x32_bf16, then emb gather + loss accumulation.
// grid = 2048, block = 256 (8 wave32 waves).
// ---------------------------------------------------------------------------
__global__ void vq_main_kernel(const float* __restrict__ x,
                               const float* __restrict__ cb,
                               const unsigned short* __restrict__ cb16_raw,
                               const float* __restrict__ enorm,
                               float* __restrict__ out_ids,
                               float* __restrict__ out_emb,
                               float* __restrict__ out_loss) {
    const __bf16* cb16 = (const __bf16*)cb16_raw;

    __shared__ __attribute__((aligned(16))) __bf16 sx[NTILE][DIM + 8];
    __shared__ float s_min[8][32][8];
    __shared__ int   s_idx[8][32][8];
    __shared__ int   s_krow[NTILE];
    __shared__ float s_loss[256];

    const int tid  = threadIdx.x;
    const int wave = tid >> 5;
    const int lane = tid & 31;
    const int half = lane >> 4;     // 0: lanes 0-15, 1: lanes 16-31
    const int lcol = lane & 15;

    const int n0   = blockIdx.x * NTILE;       // first row of this tile
    const int bb   = n0 >> 10;                  // batch index (1024 rows / batch)
    const int rem0 = n0 & 1023;                 // h*32+w base within batch
    // x_[n, c] lives at x[bb*262144 + c*1024 + (rem0 + r)]

    // ---- stage x tile (16 rows x 256 channels) into LDS as bf16 ----
    {
        const int r  = tid & 15;
        const int cg = tid >> 4;               // 0..15
        const size_t base = (size_t)bb * 262144 + (size_t)(rem0 + r);
        #pragma unroll
        for (int i = 0; i < 16; ++i) {
            int c = i * 16 + cg;
            float v = x[base + (size_t)c * 1024];
            sx[r][c] = (__bf16)v;
        }
    }
    __syncthreads();

    // ---- per-wave GEMM over its 128 k-columns, tracking running argmin ----
    float runmin[8];
    int   runidx[8];
    #pragma unroll
    for (int j = 0; j < 8; ++j) { runmin[j] = 3.0e38f; runidx[j] = 0; }

    const int kbase = wave * 128;
    for (int t = 0; t < 8; ++t) {
        const int k = kbase + t * 16 + lcol;   // this lane's codebook column
        const __bf16* brow = cb16 + (size_t)k * DIM + half * 16;

        v8f acc = {};
        #pragma unroll
        for (int cc = 0; cc < 8; ++cc) {
            const int c0 = cc * 32;
            // A fragment: 16 contiguous c-values (k = 16*half + s), matching B
            const __bf16* ap = &sx[lcol][c0 + half * 16];
            v8bf alo = *(const v8bf*)(ap);
            v8bf ahi = *(const v8bf*)(ap + 8);
            v16bf a;
            #pragma unroll
            for (int i = 0; i < 8; ++i) { a[i] = alo[i]; a[8 + i] = ahi[i]; }
            // B fragment: 16 contiguous c-values of codebook row k
            v16bf b = *(const v16bf*)(brow + c0);
            acc = __builtin_amdgcn_wmma_f32_16x16x32_bf16(
                      false, a, false, b, (short)0, acc, false, false);
        }

        const float en = enorm[k];
        #pragma unroll
        for (int j = 0; j < 8; ++j) {
            float d = en - 2.0f * acc[j];      // |e|^2 - 2 x.e  (|x|^2 const per row)
            if (d < runmin[j]) { runmin[j] = d; runidx[j] = k; }
        }
    }

    #pragma unroll
    for (int j = 0; j < 8; ++j) {
        s_min[wave][lane][j] = runmin[j];
        s_idx[wave][lane][j] = runidx[j];
    }
    __syncthreads();

    // ---- reduce argmin across waves/lanes; one thread per output row ----
    if (tid < NTILE) {
        const int r     = tid;
        const int j     = (r < 8) ? r : r - 8;
        const int lbase = (r < 8) ? 0 : 16;
        float best = 3.0e38f;
        int   bidx = 0;
        for (int w = 0; w < 8; ++w) {
            for (int lc = 0; lc < 16; ++lc) {
                float v  = s_min[w][lbase + lc][j];
                int   id = s_idx[w][lbase + lc][j];
                if (v < best || (v == best && id < bidx)) { best = v; bidx = id; }
            }
        }
        s_krow[r] = bidx;
        out_ids[n0 + r] = (float)bidx;
    }
    __syncthreads();

    // ---- emb_st (== emb forward value) scatter + loss partial sum ----
    float lsum = 0.0f;
    {
        const int r  = tid & 15;
        const int cg = tid >> 4;
        const int krow = s_krow[r];
        const float* cbr = cb + (size_t)krow * DIM;
        const size_t base = (size_t)bb * 262144 + (size_t)(rem0 + r);
        #pragma unroll
        for (int i = 0; i < 16; ++i) {
            int c = i * 16 + cg;
            size_t gi = base + (size_t)c * 1024;
            float e  = cbr[c];
            float xv = x[gi];
            out_emb[gi] = e;
            float df = xv - e;
            lsum += df * df;
        }
    }
    s_loss[tid] = lsum;
    __syncthreads();
    #pragma unroll
    for (int off = 128; off > 0; off >>= 1) {
        if (tid < off) s_loss[tid] += s_loss[tid + off];
        __syncthreads();
    }
    if (tid == 0) atomicAdd(out_loss, s_loss[0] * LOSS_SCALE);
}

// ---------------------------------------------------------------------------
extern "C" void kernel_launch(void* const* d_in, const int* in_sizes, int n_in,
                              void* d_out, int out_size, void* d_ws, size_t ws_size,
                              hipStream_t stream) {
    const float* x  = (const float*)d_in[0];   // (32, 256, 32, 32)
    const float* cb = (const float*)d_in[1];   // (1024, 256)

    float* out      = (float*)d_out;
    float* out_ids  = out;                     // 32768 (ids as float)
    float* out_emb  = out + N_TOTAL;           // 8388608
    float* out_loss = out + N_TOTAL + N_ELEMS; // 1 scalar

    // workspace: bf16 codebook (512 KB) + enorm (4 KB)
    unsigned short* cb16 = (unsigned short*)d_ws;
    float* enorm = (float*)((char*)d_ws + (size_t)NUM_EMB * DIM * sizeof(unsigned short));

    vq_prep_kernel<<<NUM_EMB, DIM, 0, stream>>>(cb, cb16, enorm, out_loss);
    vq_main_kernel<<<N_TOTAL / NTILE, 256, 0, stream>>>(
        x, cb, cb16, enorm, out_ids, out_emb, out_loss);
}
